// AutomaticCyclicalSamplerOrdinal_76355928588807
// MI455X (gfx1250) — matrix operand order repaired and checked by reading, not compiled, not run on hardware
//
#include <hip/hip_runtime.h>
#include <hip/hip_bf16.h>
#include <math.h>

typedef __attribute__((ext_vector_type(16))) _Float16 v16h;
typedef __attribute__((ext_vector_type(8)))  _Float16 v8h;
typedef __attribute__((ext_vector_type(8)))  float    v8f;
typedef __attribute__((ext_vector_type(4)))  float    v4f;

#define BB 256
#define DD 784
#define DP 800      // K padded to multiple of 32 for WMMA
#define VV 128
#define SS 2

// ---------------------------------------------------------------------------
// One-time prep
// ---------------------------------------------------------------------------
__global__ void k_wsym_f16(const float* __restrict__ W, _Float16* __restrict__ wh) {
    int i = blockIdx.x * blockDim.x + threadIdx.x;
    if (i >= DD * DP) return;
    int r = i / DP, c = i % DP;
    float v = (c < DD) ? 0.5f * (W[r * DD + c] + W[c * DD + r]) : 0.0f;
    wh[i] = (_Float16)v;
}

__global__ void k_copy_x(const int* __restrict__ xin, int* __restrict__ xcur) {
    int i = blockIdx.x * blockDim.x + threadIdx.x;
    if (i < BB * DD) xcur[i] = xin[i];
}

__global__ void k_cast_x(const int* __restrict__ x, _Float16* __restrict__ xh) {
    int i = blockIdx.x * blockDim.x + threadIdx.x;
    if (i >= BB * DP) return;
    int b = i / DP, k = i % DP;
    xh[i] = (k < DD) ? (_Float16)(float)x[b * DD + k] : (_Float16)0.0f;
}

__global__ void k_zero_acc(float* __restrict__ acc4) {   // 4 arrays of BB floats
    int i = blockIdx.x * blockDim.x + threadIdx.x;
    if (i < 4 * BB) acc4[i] = 0.0f;
}

// ---------------------------------------------------------------------------
// GEMM: g[m][n] = sum_k xh[m][k] * Wsym[k][n] + bias[n]
// Wsym symmetric -> B-fragment[k][n] = wh[n][k] (contiguous in k per lane).
// One wave32 per 16x16 output tile; K loop of 25 x v_wmma_f32_16x16x32_f16.
// ---------------------------------------------------------------------------
__global__ void k_gemm_wmma(const _Float16* __restrict__ xh,   // [BB][DP]
                            const _Float16* __restrict__ wh,   // [DD][DP]
                            const float*    __restrict__ bias, // [DD]
                            float*          __restrict__ g)    // [BB][DD]
{
    const int NT = DD / 16;                         // 49 N-tiles
    int wave = (blockIdx.x * blockDim.x + threadIdx.x) >> 5;
    int lane = threadIdx.x & 31;
    if (wave >= (BB / 16) * NT) return;
    int tm = wave / NT, tn = wave % NT;
    int m0 = tm * 16, n0 = tn * 16;
    int ml = lane & 15;                             // M (A row) / N (B col)
    int hi = lane >> 4;                             // half-wave selector

    const _Float16* arow = xh + (size_t)(m0 + ml) * DP;   // A row m
    const _Float16* brow = wh + (size_t)(n0 + ml) * DP;   // B col n == Wsym row n

    v8f acc = {};
    for (int k0 = 0; k0 < DP; k0 += 32) {
        // A fragment, contiguous k order chosen here (same as B): lanes 0-15 hold K=0..15, lanes 16-31 hold K=16..31
        v8h alo = *(const v8h*)(arow + k0 + hi * 16);
        v8h ahi = *(const v8h*)(arow + k0 + hi * 16 + 8);
        v16h a;
#pragma unroll
        for (int i = 0; i < 8; ++i) { a[i] = alo[i]; a[8 + i] = ahi[i]; }
        // B fragment, same contiguous k order as A: lanes 0-15 hold K=0..15, lanes 16-31 hold K=16..31
        v16h b = *(const v16h*)(brow + k0 + hi * 16);
        acc = __builtin_amdgcn_wmma_f32_16x16x32_f16(false, a, false, b,
                                                     (short)0, acc, false, false);
    }
    // D layout: VGPR r -> row m0 + r + 8*hi, col n0 + (lane&15)
    int n = n0 + ml;
    float bv = bias[n];
#pragma unroll
    for (int r = 0; r < 8; ++r)
        g[(size_t)(m0 + r + 8 * hi) * DD + n] = acc[r] + bv;
}

// ---------------------------------------------------------------------------
// Forward pass: one wave per (b,d). Lane l owns v = 4l..4l+3, so the 512B
// gumbel row is one nontemporal global_load_b128 per lane (single-use
// stream: NT keeps it from evicting the L2-resident W/g/x working set).
// Computes logits on the fly, argmax(logit+gumbel) (first-index tie-break,
// matching jnp.argmax), logsumexp, accumulates lp_forward and the energy
// term 0.5*xf*(g+b) per batch row.
// ---------------------------------------------------------------------------
__global__ void k_forward(const int*   __restrict__ xcur,
                          const float* __restrict__ g,
                          const float* __restrict__ bias,
                          const float* __restrict__ gum,     // this step's [BB*DD*VV]
                          const int*   __restrict__ kiter,
                          int*   __restrict__ xdel,
                          float* __restrict__ lp_fwd,
                          float* __restrict__ e_fwd)
{
    int gid  = blockIdx.x * blockDim.x + threadIdx.x;
    int row  = gid >> 5;
    int lane = gid & 31;
    if (row >= BB * DD) return;
    int b = row / DD, d = row % DD;

    bool big    = ((kiter[0] % 10) == 0);
    float bal   = big ? 0.9f : 0.6f;
    float inv2s = 0.5f / (big ? 0.5f : 0.1f);

    float xf = (float)xcur[row];
    float gv = g[row];
    const v4f* gp4 = (const v4f*)(gum + (size_t)row * VV) + lane;
    v4f gn = __builtin_nontemporal_load(gp4);       // global_load_b128, TH=NT

    float l[4];
    float lmax = -1e30f;
    float bestv = -1e30f, bestlog = 0.0f;
    int   besti = 0;
#pragma unroll
    for (int j = 0; j < 4; ++j) {
        int v = 4 * lane + j;
        float diff = (float)v - xf;
        float lg = gv * diff * bal - diff * diff * inv2s;
        l[j] = lg;
        lmax = fmaxf(lmax, lg);
        float pert = lg + gn[j];
        if (pert > bestv) { bestv = pert; besti = v; bestlog = lg; }
    }
    for (int off = 16; off > 0; off >>= 1) {
        float ov = __shfl_xor(bestv, off, 32);
        int   oi = __shfl_xor(besti, off, 32);
        float ol = __shfl_xor(bestlog, off, 32);
        if (ov > bestv || (ov == bestv && oi < besti)) {
            bestv = ov; besti = oi; bestlog = ol;
        }
        lmax = fmaxf(lmax, __shfl_xor(lmax, off, 32));
    }
    float s = 0.0f;
#pragma unroll
    for (int j = 0; j < 4; ++j) s += expf(l[j] - lmax);
    for (int off = 16; off > 0; off >>= 1) s += __shfl_xor(s, off, 32);

    if (lane == 0) {
        xdel[row] = besti;
        float lse = lmax + logf(s);
        atomicAdd(&lp_fwd[b], bestlog - lse);
        atomicAdd(&e_fwd[b], 0.5f * xf * (gv + bias[d]));
    }
}

// ---------------------------------------------------------------------------
// Reverse pass: same logits shape for the proposal; no gumbel read. Needs
// log_softmax evaluated at the *current* x.
// ---------------------------------------------------------------------------
__global__ void k_reverse(const int*   __restrict__ xcur,
                          const int*   __restrict__ xdel,
                          const float* __restrict__ gd,
                          const float* __restrict__ bias,
                          const int*   __restrict__ kiter,
                          float* __restrict__ lp_rev,
                          float* __restrict__ e_rev)
{
    int gid  = blockIdx.x * blockDim.x + threadIdx.x;
    int row  = gid >> 5;
    int lane = gid & 31;
    if (row >= BB * DD) return;
    int b = row / DD, d = row % DD;

    bool big    = ((kiter[0] % 10) == 0);
    float bal   = big ? 0.9f : 0.6f;
    float inv2s = 0.5f / (big ? 0.5f : 0.1f);

    int   xc  = xcur[row];
    float xdf = (float)xdel[row];
    float gv  = gd[row];

    float l[4];
    float lmax = -1e30f, at = -1e30f;
#pragma unroll
    for (int j = 0; j < 4; ++j) {
        int v = 4 * lane + j;
        float diff = (float)v - xdf;
        float lg = gv * diff * bal - diff * diff * inv2s;
        l[j] = lg;
        lmax = fmaxf(lmax, lg);
        if (v == xc) at = lg;
    }
    for (int off = 16; off > 0; off >>= 1) {
        lmax = fmaxf(lmax, __shfl_xor(lmax, off, 32));
        at   = fmaxf(at,   __shfl_xor(at,   off, 32));
    }
    float s = 0.0f;
#pragma unroll
    for (int j = 0; j < 4; ++j) s += expf(l[j] - lmax);
    for (int off = 16; off > 0; off >>= 1) s += __shfl_xor(s, off, 32);

    if (lane == 0) {
        float lse = lmax + logf(s);
        atomicAdd(&lp_rev[b], at - lse);
        atomicAdd(&e_rev[b], 0.5f * xdf * (gv + bias[d]));
    }
}

// ---------------------------------------------------------------------------
// MH accept + state update; last step also writes d_out = [x_out (as float),
// la] per the reference's (x_out, las[-1]) return order.
// ---------------------------------------------------------------------------
__global__ void k_accept(const float* __restrict__ e_fwd, const float* __restrict__ e_rev,
                         const float* __restrict__ lp_fwd, const float* __restrict__ lp_rev,
                         const float* __restrict__ u,       // this step's [BB]
                         int* __restrict__ xcur, const int* __restrict__ xdel,
                         float* __restrict__ out, int write_out)
{
    int b = blockIdx.x;
    __shared__ int   s_acc;
    __shared__ float s_la;
    if (threadIdx.x == 0) {
        float la = (e_rev[b] - e_fwd[b]) + lp_rev[b] - lp_fwd[b];
        s_la  = la;
        s_acc = (expf(la) > u[b]) ? 1 : 0;
    }
    __syncthreads();
    int acc = s_acc;
    for (int d = threadIdx.x; d < DD; d += blockDim.x) {
        size_t i = (size_t)b * DD + d;
        int xn = acc ? xdel[i] : xcur[i];
        xcur[i] = xn;
        if (write_out) out[i] = (float)xn;
    }
    if (write_out && threadIdx.x == 0) out[(size_t)BB * DD + b] = s_la;
}

// ---------------------------------------------------------------------------
extern "C" void kernel_launch(void* const* d_in, const int* in_sizes, int n_in,
                              void* d_out, int out_size, void* d_ws, size_t ws_size,
                              hipStream_t stream) {
    const int*   x_in  = (const int*)  d_in[0];
    const float* W     = (const float*)d_in[1];
    const float* bias  = (const float*)d_in[2];
    const float* gum   = (const float*)d_in[3];   // [S][B][D][V]
    const float* u     = (const float*)d_in[4];   // [S][B]
    const int*   kiter = (const int*)  d_in[5];
    float* out = (float*)d_out;

    // workspace carve-up (256B aligned)
    char* ws = (char*)d_ws;
    size_t off = 0;
    auto alloc = [&](size_t bytes) {
        char* p = ws + off;
        off = (off + bytes + 255) & ~(size_t)255;
        return p;
    };
    _Float16* wh   = (_Float16*)alloc((size_t)DD * DP * 2);
    _Float16* xh   = (_Float16*)alloc((size_t)BB * DP * 2);
    float*    g    = (float*)   alloc((size_t)BB * DD * 4);
    float*    gd   = (float*)   alloc((size_t)BB * DD * 4);
    int*      xcur = (int*)     alloc((size_t)BB * DD * 4);
    int*      xdel = (int*)     alloc((size_t)BB * DD * 4);
    float*    acc4 = (float*)   alloc((size_t)4 * BB * 4);  // lp_f, lp_r, e_f, e_r
    float* lp_f = acc4 + 0 * BB;
    float* lp_r = acc4 + 1 * BB;
    float* e_f  = acc4 + 2 * BB;
    float* e_r  = acc4 + 3 * BB;

    const int T = 256;
    int gemm_waves  = (BB / 16) * (DD / 16);            // 784
    int gemm_blocks = (gemm_waves * 32 + T - 1) / T;    // 98
    int row_blocks  = (BB * DD * 32 + T - 1) / T;       // 25088

    k_wsym_f16<<<(DD * DP + T - 1) / T, T, 0, stream>>>(W, wh);
    k_copy_x  <<<(BB * DD + T - 1) / T, T, 0, stream>>>(x_in, xcur);

    for (int s = 0; s < SS; ++s) {
        const float* gum_s = gum + (size_t)s * BB * DD * VV;
        const float* u_s   = u   + (size_t)s * BB;

        k_zero_acc<<<(4 * BB + T - 1) / T, T, 0, stream>>>(acc4);

        k_cast_x   <<<(BB * DP + T - 1) / T, T, 0, stream>>>(xcur, xh);
        k_gemm_wmma<<<gemm_blocks, T, 0, stream>>>(xh, wh, bias, g);
        k_forward  <<<row_blocks, T, 0, stream>>>(xcur, g, bias, gum_s, kiter,
                                                  xdel, lp_f, e_f);

        k_cast_x   <<<(BB * DP + T - 1) / T, T, 0, stream>>>(xdel, xh);
        k_gemm_wmma<<<gemm_blocks, T, 0, stream>>>(xh, wh, bias, gd);
        k_reverse  <<<row_blocks, T, 0, stream>>>(xcur, xdel, gd, bias, kiter,
                                                  lp_r, e_r);

        k_accept<<<BB, T, 0, stream>>>(e_f, e_r, lp_f, lp_r, u_s,
                                       xcur, xdel, out, (s == SS - 1) ? 1 : 0);
    }
}
